// Attention_v1_13735305413199
// MI455X (gfx1250) — matrix operand order repaired and checked by reading, not compiled, not run on hardware
//
#include <hip/hip_runtime.h>
#include <math.h>

// ---------------- problem constants (from setup_inputs) ----------------
#define B_SZ 4
#define CDIM 192
#define HH 128
#define WW 128
#define NTOT (HH * WW)          // 16384
#define HEADS 8
#define DH (CDIM / HEADS)       // 24

// ---------------- WMMA types ----------------
typedef __bf16 bhalf;
typedef bhalf v16bf __attribute__((ext_vector_type(16)));
typedef float v8f  __attribute__((ext_vector_type(8)));

union BF16Frag { v16bf v; unsigned int u[8]; };

// gfx1250 async-load-to-LDS path (guarded; falls back to VGPR staging)
#if __has_builtin(__builtin_amdgcn_global_load_async_to_lds_b128) && \
    __has_builtin(__builtin_amdgcn_s_wait_asynccnt)
#define HAVE_GFX1250_ASYNC 1
#else
#define HAVE_GFX1250_ASYNC 0
#endif

// async builtin takes (int4 addrspace(1)*, int4 addrspace(3)*, imm, imm)
typedef int v4i __attribute__((ext_vector_type(4)));
typedef __attribute__((address_space(1))) v4i* gptr_v4i;
typedef __attribute__((address_space(3))) v4i* lptr_v4i;

// f32 -> packed bf16x2 using V_FMA_MIXLO_BF16 / V_FMA_MIXHI_BF16
// (gfx1250 ISA 15.10, opcodes 62/63): d = cvt(fma(x, 1.0, 0)) per half.
__device__ __forceinline__ unsigned int packbf(float lo, float hi) {
  unsigned int d = 0;
  asm("v_fma_mixlo_bf16 %0, %1, 1.0, 0\n\t"
      "v_fma_mixhi_bf16 %0, %2, 1.0, 0"
      : "+v"(d) : "v"(lo), "v"(hi));
  return d;
}

// =====================================================================
// 1) Conv1x1 GEMM: Y[b,o,n] = sum_c W[o,c] * X[b,c,n], c in [0,192).
//    Block = 4 waves = 64(M) x 64(N) tile. Per K=32 chunk: the block
//    cooperatively stages the 32x64 X chunk as packed bf16 pairs in LDS
//    in FRAGMENT-MAJOR order [col][pair] so each lane's B fragment is
//    two 4-dword, 16B-aligned runs (+0..3, +8..11) -> 2x ds_load_b128, no reg shuffle.
// =====================================================================
#define GPIT 20  // dwords per column: 16 pairs + pad; (col*20+half*4)*4 is
                 // 16B aligned; lanes 0-15 start banks 20*row%64, spacing 4

__global__ __launch_bounds__(128) void k_gemm_wx(
    const float* __restrict__ W, const float* __restrict__ X,
    float* __restrict__ Y, int O, long xbStride)
{
  __shared__ alignas(16) unsigned int ldsB[64 * GPIT];   // [col][pair]
  const int tid  = threadIdx.x;
  const int lane = tid & 31;
  const int wave = tid >> 5;
  const int numMb = O >> 6;                    // blocks of 64 output rows
  const int numNs = NTOT >> 6;                 // 256 strips of 64 columns
  const long perB = (long)numMb * numNs;
  const int b  = (int)(blockIdx.x / perB);
  const int rm = (int)(blockIdx.x % perB);
  const int mb = rm / numNs;
  const int ns = rm % numNs;
  const int ot = mb * 4 + wave;                // this wave's 16-row M tile
  const int row  = lane & 15;
  const int half = lane >> 4;

  const float* Xb = X + (long)b * xbStride + (long)ns * 64;
  float* Yb = Y + ((long)b * O + ot * 16) * NTOT + (long)ns * 64;

  const int sn = tid & 63;                     // staging: n within strip
  const int sp = tid >> 6;                     // staging: base pair (0/1)

  v8f acc[4];
#pragma unroll
  for (int s = 0; s < 4; ++s)
#pragma unroll
    for (int i = 0; i < 8; ++i) acc[s][i] = 0.f;

  for (int kc = 0; kc < 6; ++kc) {
    __syncthreads();                           // previous chunk consumed
    // ---- stage 16 c-pairs x 64 n (coalesced global), pack once per block
#pragma unroll
    for (int j = 0; j < 8; ++j) {
      const int pr = sp + 2 * j;               // pair 0..15
      const int c  = kc * 32 + 2 * pr;
      const float lo = Xb[(long)c * NTOT + sn];
      const float hi = Xb[(long)(c + 1) * NTOT + sn];
      ldsB[sn * GPIT + pr] = packbf(lo, hi);
    }
    __syncthreads();

    // ---- A fragment from W (tiny, L2-hot). 16-bit A layout:
    // lane<16: row M=lane, K = klo+{0..7} in v0..3, klo+16+{0..7} in v4..7
    BF16Frag A;
    const float* wr = W + (ot * 16 + row) * CDIM + kc * 32 + half * 8;
    float4 a0 = *(const float4*)(wr + 0);
    float4 a1 = *(const float4*)(wr + 4);
    float4 a2 = *(const float4*)(wr + 16);
    float4 a3 = *(const float4*)(wr + 20);
    A.u[0] = packbf(a0.x, a0.y); A.u[1] = packbf(a0.z, a0.w);
    A.u[2] = packbf(a1.x, a1.y); A.u[3] = packbf(a1.z, a1.w);
    A.u[4] = packbf(a2.x, a2.y); A.u[5] = packbf(a2.z, a2.w);
    A.u[6] = packbf(a3.x, a3.y); A.u[7] = packbf(a3.z, a3.w);

    // ---- B fragments: pairs half*4+{0..3} and 8+half*4+{0..3} per lane
#pragma unroll
    for (int s = 0; s < 4; ++s) {
      BF16Frag Bt;
      const unsigned int* bp = ldsB + (s * 16 + row) * GPIT + half * 4;
#pragma unroll
      for (int i = 0; i < 4; ++i) { Bt.u[i] = bp[i]; Bt.u[4 + i] = bp[8 + i]; }
      acc[s] = __builtin_amdgcn_wmma_f32_16x16x32_bf16(
          false, A.v, false, Bt.v, (short)0, acc[s], false, false);
    }
  }

  // D layout: VGPR r -> M = r + 8*half, N = lane&15
#pragma unroll
  for (int s = 0; s < 4; ++s)
#pragma unroll
    for (int r = 0; r < 8; ++r)
      Yb[(long)(r + half * 8) * NTOT + s * 16 + row] = acc[s][r];
}

// =====================================================================
// 2) Depthwise 3x3 conv, SAME zero padding. [B, 576, 128, 128]
// =====================================================================
__global__ __launch_bounds__(256) void k_dwconv(
    const float* __restrict__ in, const float* __restrict__ wdw,
    float* __restrict__ out)
{
  const long idx = (long)blockIdx.x * 256 + threadIdx.x;
  const int w = (int)(idx & (WW - 1));
  const int h = (int)((idx >> 7) & (HH - 1));
  const long ch = idx >> 14;                   // b*576 + c
  const int c = (int)(ch % (3 * CDIM));
  const float* wp = wdw + c * 9;
  const float* ip = in + (ch << 14);
  float s = 0.f;
#pragma unroll
  for (int dy = -1; dy <= 1; ++dy) {
    const int hy = h + dy;
    if (hy < 0 || hy >= HH) continue;
#pragma unroll
    for (int dx = -1; dx <= 1; ++dx) {
      const int wx = w + dx;
      if (wx < 0 || wx >= WW) continue;
      s += wp[(dy + 1) * 3 + (dx + 1)] * ip[hy * WW + wx];
    }
  }
  out[idx] = s;
}

// =====================================================================
// 3) L2 norms of q/k rows (channels 0..383 of qkv_dw), over n = 16384
// =====================================================================
__global__ __launch_bounds__(256) void k_rownorm(
    const float* __restrict__ qkv, float* __restrict__ norms)
{
  const int rid = blockIdx.x;                  // B * 384
  const int b = rid / (2 * CDIM);
  const int c = rid % (2 * CDIM);
  const float* p = qkv + ((long)b * 3 * CDIM + c) * NTOT;
  float s = 0.f;
  for (int i = threadIdx.x; i < NTOT; i += 256) { float v = p[i]; s += v * v; }
  __shared__ float red[256];
  red[threadIdx.x] = s; __syncthreads();
  for (int off = 128; off; off >>= 1) {
    if (threadIdx.x < (unsigned)off) red[threadIdx.x] += red[threadIdx.x + off];
    __syncthreads();
  }
  if (threadIdx.x == 0) norms[rid] = fmaxf(sqrtf(red[0]), 1e-12f);
}

// =====================================================================
// 4) Attention scores: one block per (b, head, d1) row; fused softmax.
// =====================================================================
__global__ __launch_bounds__(256) void k_attn_score(
    const float* __restrict__ qkv, const float* __restrict__ norms,
    const float* __restrict__ temp, float* __restrict__ attn)
{
  const int id = blockIdx.x;                   // B*HEADS*DH
  const int d1 = id % DH;
  const int hd = (id / DH) % HEADS;
  const int b  = id / (DH * HEADS);
  const float* qrow = qkv + ((long)b * 3 * CDIM + hd * DH + d1) * NTOT;
  const float* kmat = qkv + ((long)b * 3 * CDIM + CDIM + hd * DH) * NTOT;

  float acc[DH];
#pragma unroll
  for (int j = 0; j < DH; ++j) acc[j] = 0.f;
  for (int n = threadIdx.x; n < NTOT; n += 256) {
    const float q = qrow[n];
#pragma unroll
    for (int j = 0; j < DH; ++j) acc[j] += q * kmat[(long)j * NTOT + n];
  }

  const int lane = threadIdx.x & 31;
  const int wave = threadIdx.x >> 5;
#pragma unroll
  for (int j = 0; j < DH; ++j)
    for (int off = 16; off; off >>= 1)
      acc[j] += __shfl_down(acc[j], off);

  __shared__ float wred[8][DH];
  __shared__ float srow[DH];
  if (lane == 0)
    for (int j = 0; j < DH; ++j) wred[wave][j] = acc[j];
  __syncthreads();
  if (threadIdx.x < DH) {
    float s = 0.f;
    for (int wv = 0; wv < 8; ++wv) s += wred[wv][threadIdx.x];
    const float qn = norms[b * 2 * CDIM + hd * DH + d1];
    const float kn = norms[b * 2 * CDIM + CDIM + hd * DH + threadIdx.x];
    srow[threadIdx.x] = s / (qn * kn) * temp[hd];
  }
  __syncthreads();
  if (threadIdx.x == 0) {
    float m = srow[0];
    for (int j = 1; j < DH; ++j) m = fmaxf(m, srow[j]);
    float e[DH]; float sum = 0.f;
    for (int j = 0; j < DH; ++j) { e[j] = __expf(srow[j] - m); sum += e[j]; }
    const float inv = 1.f / sum;
    float* ap = attn + (long)id * DH;
    for (int j = 0; j < DH; ++j) ap[j] = e[j] * inv;
  }
}

// =====================================================================
// 5) out[b, head*24+d1, n] = sum_d2 attn[d1,d2] * v[d2, n]
// =====================================================================
__global__ __launch_bounds__(256) void k_attn_apply(
    const float* __restrict__ attn, const float* __restrict__ qkv,
    float* __restrict__ out)
{
  const int id = blockIdx.x;                   // B*HEADS*64
  const int nc = id & 63;
  const int hd = (id >> 6) & 7;
  const int b  = id >> 9;
  __shared__ float a[DH * DH];
  for (int i = threadIdx.x; i < DH * DH; i += 256)
    a[i] = attn[(long)(b * HEADS + hd) * DH * DH + i];
  __syncthreads();
  const int n = nc * 256 + threadIdx.x;
  const float* vb = qkv + ((long)b * 3 * CDIM + 2 * CDIM + hd * DH) * NTOT + n;
  float v[DH];
#pragma unroll
  for (int j = 0; j < DH; ++j) v[j] = vb[(long)j * NTOT];
  float* ob = out + ((long)b * CDIM + hd * DH) * NTOT + n;
  for (int d1 = 0; d1 < DH; ++d1) {
    float s = 0.f;
#pragma unroll
    for (int j = 0; j < DH; ++j) s += a[d1 * DH + j] * v[j];
    ob[(long)d1 * NTOT] = s;
  }
}

// =====================================================================
// 6) Per-row (b,c,h) softmax stats over w (128): (max, sum of exp)
// =====================================================================
__global__ __launch_bounds__(128) void k_rowstats(
    const float* __restrict__ x, float2* __restrict__ st)
{
  const long row = blockIdx.x;                 // B*192*128 rows
  const float v = x[row * WW + threadIdx.x];
  __shared__ float red[128];
  red[threadIdx.x] = v; __syncthreads();
  for (int off = 64; off; off >>= 1) {
    if (threadIdx.x < (unsigned)off)
      red[threadIdx.x] = fmaxf(red[threadIdx.x], red[threadIdx.x + off]);
    __syncthreads();
  }
  const float m = red[0];
  __syncthreads();
  const float e = __expf(v - m);
  red[threadIdx.x] = e; __syncthreads();
  for (int off = 64; off; off >>= 1) {
    if (threadIdx.x < (unsigned)off) red[threadIdx.x] += red[threadIdx.x + off];
    __syncthreads();
  }
  if (threadIdx.x == 0) st[row] = make_float2(m, red[0]);
}

// =====================================================================
// 7) Cross matmul: dst = softmax(o1)@o2 + softmax(o2)@o1, per (b,c) 128x128.
//    Stage f32 via async-to-LDS; one cooperative conversion pass builds
//    packed-bf16 fragment-major arrays; compute loop is b128 ds loads +
//    WMMA only.
// =====================================================================
#define SP_PITCH 68    // S: [row][K-pair], 272*h is 16B aligned -> 2x b128
#define BP_PITCH 68    // B: [col][K-pair], 272*col 16B aligned -> 2x b128

__global__ __launch_bounds__(256) void k_cross(
    const float* __restrict__ o1, const float* __restrict__ o2,
    const float2* __restrict__ st1, const float2* __restrict__ st2,
    float* __restrict__ dst)
{
  extern __shared__ float smem[];
  float* M1 = smem;                                   // 16384 f32
  float* M2 = smem + NTOT;                            // 16384 f32
  unsigned int* Sp1 = (unsigned int*)(smem + 2 * NTOT);
  unsigned int* Sp2 = Sp1 + HH * SP_PITCH;            // 8704 dwords each
  unsigned int* Bp1 = Sp2 + HH * SP_PITCH;
  unsigned int* Bp2 = Bp1 + WW * BP_PITCH;            // 8704 dwords each

  const int bc = blockIdx.x;       // b*192 + c
  const float4* g1 = (const float4*)(o1 + (long)bc * NTOT);
  const float4* g2 = (const float4*)(o2 + (long)bc * NTOT);
#if HAVE_GFX1250_ASYNC
  {
    float4* l1 = (float4*)M1;
    float4* l2 = (float4*)M2;
    for (int i = threadIdx.x; i < NTOT / 4; i += 256) {
      __builtin_amdgcn_global_load_async_to_lds_b128(
          (gptr_v4i)(uintptr_t)(g1 + i),
          (lptr_v4i)(unsigned int)(uintptr_t)(l1 + i), 0, 0);
      __builtin_amdgcn_global_load_async_to_lds_b128(
          (gptr_v4i)(uintptr_t)(g2 + i),
          (lptr_v4i)(unsigned int)(uintptr_t)(l2 + i), 0, 0);
    }
    __builtin_amdgcn_s_wait_asynccnt(0);
  }
#else
  for (int i = threadIdx.x; i < NTOT / 4; i += 256) {
    ((float4*)M1)[i] = g1[i];
    ((float4*)M2)[i] = g2[i];
  }
#endif
  __syncthreads();

  // ---- conversion pre-pass (once per block) ----
  // B arrays, fragment-major: Bp[col][pair(w,w+1)]
  for (int i = threadIdx.x; i < 64 * WW; i += 256) {
    const int pr = i >> 7, col = i & 127;     // col consecutive per lane
    Bp1[col * BP_PITCH + pr] =
        packbf(M1[(2 * pr) * WW + col], M1[(2 * pr + 1) * WW + col]);
    Bp2[col * BP_PITCH + pr] =
        packbf(M2[(2 * pr) * WW + col], M2[(2 * pr + 1) * WW + col]);
  }
  // S arrays: softmax probabilities, Sp[row h][pair(w,w+1)]
  for (int i = threadIdx.x; i < HH * 64; i += 256) {
    const int hh = i >> 6, pj = i & 63;
    const float2 t1 = st1[(long)bc * HH + hh];
    const float2 t2 = st2[(long)bc * HH + hh];
    const float r1 = 1.f / t1.y;
    const float r2 = 1.f / t2.y;
    const int w = 2 * pj;
    Sp1[hh * SP_PITCH + pj] = packbf(__expf(M1[hh * WW + w] - t1.x) * r1,
                                     __expf(M1[hh * WW + w + 1] - t1.x) * r1);
    Sp2[hh * SP_PITCH + pj] = packbf(__expf(M2[hh * WW + w] - t2.x) * r2,
                                     __expf(M2[hh * WW + w + 1] - t2.x) * r2);
  }
  __syncthreads();

  const int wave = threadIdx.x >> 5;
  const int lane = threadIdx.x & 31;
  const int row  = lane & 15;
  const int half = lane >> 4;
  const int h    = wave * 16 + row;            // A-row this lane owns

  v8f acc[8];
#pragma unroll
  for (int s = 0; s < 8; ++s)
#pragma unroll
    for (int i = 0; i < 8; ++i) acc[s][i] = 0.f;

#pragma unroll
  for (int term = 0; term < 2; ++term) {
    const unsigned int* Sp = term ? Sp2 : Sp1;
    const unsigned int* Bp = term ? Bp1 : Bp2;

    for (int kc = 0; kc < 4; ++kc) {
      // A fragment: pairs klo/2 + {0..3} in u[0..3], +8+{0..3} in u[4..7]
      BF16Frag A;
      const unsigned int* ap = Sp + h * SP_PITCH + kc * 16 + half * 4;
#pragma unroll
      for (int i = 0; i < 4; ++i) { A.u[i] = ap[i]; A.u[4 + i] = ap[8 + i]; }

#pragma unroll
      for (int s = 0; s < 8; ++s) {
        BF16Frag Bt;
        const unsigned int* bp =
            Bp + (s * 16 + row) * BP_PITCH + kc * 16 + half * 4;
#pragma unroll
        for (int i = 0; i < 4; ++i) { Bt.u[i] = bp[i]; Bt.u[4 + i] = bp[8 + i]; }
        acc[s] = __builtin_amdgcn_wmma_f32_16x16x32_bf16(
            false, A.v, false, Bt.v, (short)0, acc[s], false, false);
      }
    }
  }

  float* d = dst + (long)bc * NTOT + (long)(wave * 16) * WW;
#pragma unroll
  for (int s = 0; s < 8; ++s)
#pragma unroll
    for (int r = 0; r < 8; ++r)
      d[(r + half * 8) * WW + s * 16 + row] = acc[s][r];
}

// =====================================================================
// Launch
// =====================================================================
extern "C" void kernel_launch(void* const* d_in, const int* in_sizes, int n_in,
                              void* d_out, int out_size, void* d_ws, size_t ws_size,
                              hipStream_t stream)
{
  (void)in_sizes; (void)n_in; (void)out_size; (void)ws_size;
  const float* x      = (const float*)d_in[0];
  const float* w_in   = (const float*)d_in[1];
  const float* w_qkv1 = (const float*)d_in[2];
  const float* w_dw1  = (const float*)d_in[3];
  const float* w_qkv2 = (const float*)d_in[4];
  const float* w_dw2  = (const float*)d_in[5];
  const float* temp   = (const float*)d_in[6];
  const float* w_out  = (const float*)d_in[7];
  float* out = (float*)d_out;

  float* ws = (float*)d_ws;
  const long SZ_XP  = (long)B_SZ * 2 * CDIM * NTOT;   // 25,165,824
  const long SZ_QKV = (long)B_SZ * 3 * CDIM * NTOT;   // 37,748,736
  const long SZ_OUT = (long)B_SZ * CDIM * NTOT;       // 12,582,912
  float* xp    = ws;
  float* raw   = xp   + SZ_XP;
  float* dwb   = raw  + SZ_QKV;
  float* out1  = dwb  + SZ_QKV;
  float* out2  = out1 + SZ_OUT;
  float* cross = out2 + SZ_OUT;
  float* norms = cross + SZ_OUT;
  float* attn  = norms + (long)B_SZ * 2 * CDIM;
  float2* st1  = (float2*)(attn + (long)B_SZ * HEADS * DH * DH);
  float2* st2  = st1 + (long)B_SZ * CDIM * HH;

  const int gGemmIn  = B_SZ * ((2 * CDIM) / 64) * (NTOT / 64);  // 6144
  const int gGemmQkv = B_SZ * ((3 * CDIM) / 64) * (NTOT / 64);  // 9216
  const int gGemmOut = B_SZ * (CDIM / 64) * (NTOT / 64);        // 3072
  const int gDw      = (int)((long)B_SZ * 3 * CDIM * NTOT / 256);

  const int crossLds =
      (2 * NTOT + 2 * HH * SP_PITCH + 2 * WW * BP_PITCH) * (int)sizeof(float);

  // 1) xp = w_in @ x
  k_gemm_wx<<<gGemmIn, 128, 0, stream>>>(w_in, x, xp, 2 * CDIM,
                                         (long)CDIM * NTOT);

  for (int branch = 0; branch < 2; ++branch) {
    const float* wq = branch ? w_qkv2 : w_qkv1;
    const float* wd = branch ? w_dw2  : w_dw1;
    float* obuf     = branch ? out2   : out1;
    // 2) qkv = w_qkv @ x_branch   (slice of xp, batch stride 384*N)
    k_gemm_wx<<<gGemmQkv, 128, 0, stream>>>(
        wq, xp + (long)branch * CDIM * NTOT, raw, 3 * CDIM,
        (long)2 * CDIM * NTOT);
    // 3) depthwise 3x3
    k_dwconv<<<gDw, 256, 0, stream>>>(raw, wd, dwb);
    // 4) q/k row norms
    k_rownorm<<<B_SZ * 2 * CDIM, 256, 0, stream>>>(dwb, norms);
    // 5) 24x24 scores + softmax
    k_attn_score<<<B_SZ * HEADS * DH, 256, 0, stream>>>(dwb, norms, temp, attn);
    // 6) attn @ v
    k_attn_apply<<<B_SZ * HEADS * 64, 256, 0, stream>>>(attn, dwb, obuf);
  }

  // 7) softmax row stats over W
  k_rowstats<<<B_SZ * CDIM * HH, 128, 0, stream>>>(out1, st1);
  k_rowstats<<<B_SZ * CDIM * HH, 128, 0, stream>>>(out2, st2);
  // 8) cross = s1@out2 + s2@out1 (WMMA, ~264KB LDS per block, async staging)
  k_cross<<<B_SZ * CDIM, 256, crossLds, stream>>>(out1, out2, st1, st2, cross);
  // 9) final conv1x1
  k_gemm_wx<<<gGemmOut, 128, 0, stream>>>(w_out, cross, out, CDIM,
                                          (long)CDIM * NTOT);
}
